// Attention_Sparse_45732811768357
// MI455X (gfx1250) — matrix operand order repaired and checked by reading, not compiled, not run on hardware
//
#include <hip/hip_runtime.h>

// ---------------------------------------------------------------------------
// Sparse batched attention for MI455X (gfx1250, wave32, WMMA bf16 16x16x32).
//
// Pipeline (all on `stream`, scratch carved from d_ws):
//   1. x, context  -> bf16 copies (WMMA operand precision; f32 accumulate)
//   2. W_qkv slices + W_out -> bf16, transposed to K-major (N x K) so both
//      WMMA operands are contiguous 16B loads
//   3. Software-pipelined GEMM (16x64 tile / wave, double-buffered frags):
//      Q -> [H][NX][64], K -> [H][NC][64], V -> transposed [H][64][NC]
//   4. cmeta[c] = {mask? frag_mask : -1e30, bits(batch_context)}; batch range
//      table via binary search (batch_context sorted) -> each q-tile only
//      scans its own batch's contiguous context range (~16x fewer FLOPs)
//   5. Flash attention per (head, 16-query) wave: S^T = K_tile @ Q^T so the
//      online-softmax stats are per-lane and P^T feeds the O^T WMMA B operand
//      with a single shfl_xor(16) lane-pair exchange (no LDS transpose).
//      K fragments for tile i+1 load right after tile i's S-WMMAs; V
//      fragments for tile i+1 load at the loop bottom (after tile i's
//      O-WMMAs release the registers) -> every load has ~1 iteration of
//      latency cover. Loop bounds readfirstlane'd to SGPRs (scalar control,
//      EXEC all-1s for WMMA).
//   6. Final GEMM: O(bf16) @ W_out^T + b_out -> f32 d_out.
//
// WMMA register-layout assumptions (CDNA5 ISA):
//   A 16x32 bf16 : lanes 0-15 row M=l, K=0..15 contiguous; lanes 16-31 K=16..31 (this code's order)
//   B 32x16 bf16 : lanes 0-15 col N=l, K=0..15 contiguous; lanes 16-31 K=16..31 (this code's order, same as A)
//   C/D 16x16 f32: VGPR r, lanes 0-15 -> M=r, N=l; lanes 16-31 -> M=r+8
// ---------------------------------------------------------------------------

constexpr int NX = 2048, NC = 8192, BATCHES = 16, DIM = 512, H = 8, DH = 64;

typedef __attribute__((ext_vector_type(16))) __bf16 v16bf;
typedef __attribute__((ext_vector_type(8)))  float  v8f;

union BFrag { v16bf v; uint4 u[2]; };

__device__ __forceinline__ __bf16 f2bf(float f) {
  unsigned u = __builtin_bit_cast(unsigned, f);
  u += 0x7FFFu + ((u >> 16) & 1u);           // round-to-nearest-even
  unsigned short s = (unsigned short)(u >> 16);
  return __builtin_bit_cast(__bf16, s);
}

// A-operand fragment: row-major matrix row, 32-wide K chunk at `row`
__device__ __forceinline__ void load_afrag(BFrag& f, const __bf16* row, int hi) {
  const __bf16* p = row + (hi ? 16 : 0);
  f.u[0] = *(const uint4*)(p);        // K = 0..7   (+16 for hi)
  f.u[1] = *(const uint4*)(p + 8);    // K = 8..15  (+16 for hi)
}
// B-operand fragment: K-major (N x K) matrix row, 32-wide K chunk at `row`
__device__ __forceinline__ void load_bfrag(BFrag& f, const __bf16* row, int hi) {
  const __bf16* p = row + (hi ? 16 : 0);
  f.u[0] = *(const uint4*)(p);        // K = 0..7   (+16 for hi)
  f.u[1] = *(const uint4*)(p + 8);    // K = 8..15  (+16 for hi)
}

__device__ __forceinline__ v8f wmma_bf16(const BFrag& a, const BFrag& b, v8f c) {
  return __builtin_amdgcn_wmma_f32_16x16x32_bf16(false, a.v, false, b.v,
                                                 (short)0, c, false, false);
}

// ---------------------------------------------------------------------------
// Prep kernels
// ---------------------------------------------------------------------------
union BF4 { __bf16 b[4]; uint2 u; };

__global__ void cvt_bf16_kernel(const float* __restrict__ in,
                                __bf16* __restrict__ out, int n4) {
  int i = blockIdx.x * blockDim.x + threadIdx.x;
  if (i >= n4) return;
  const float4 f = ((const float4*)in)[i];
  BF4 o;
  o.b[0] = f2bf(f.x); o.b[1] = f2bf(f.y);
  o.b[2] = f2bf(f.z); o.b[3] = f2bf(f.w);
  ((uint2*)out)[i] = o.u;
}

// Wt[n*512 + k] = W[k*Nfull + coff + n]  (512x512 bf16 K-major slice)
__global__ void transpose_w_kernel(const float* __restrict__ W, int Nfull,
                                   int coff, __bf16* __restrict__ Wt) {
  int idx = blockIdx.x * blockDim.x + threadIdx.x;
  if (idx >= 512 * 512) return;
  int n = idx >> 9, k = idx & 511;
  Wt[idx] = f2bf(W[(size_t)k * Nfull + coff + n]);
}

__global__ void prep_cmeta_kernel(const int* __restrict__ bc,
                                  const unsigned char* __restrict__ mask,
                                  const float* __restrict__ fm,
                                  float2* __restrict__ cmeta, int n) {
  int i = blockIdx.x * blockDim.x + threadIdx.x;
  if (i < n) {
    float fv = mask[i] ? fm[i] : -1e30f;
    cmeta[i] = make_float2(fv, __int_as_float(bc[i]));
  }
}

// range[b] = lower_bound(batch_context, b); batch_context is sorted.
__global__ void batch_ranges_kernel(const int* __restrict__ bc,
                                    int* __restrict__ range) {
  int b = threadIdx.x;
  if (b > BATCHES) return;
  int lo = 0, hires = NC;
  while (lo < hires) {
    int mid = (lo + hires) >> 1;
    if (bc[mid] < b) lo = mid + 1; else hires = mid;
  }
  range[b] = lo;
}

// ---------------------------------------------------------------------------
// Generic WMMA GEMM: C[M x 512] = A[M x K] @ Wt[512 x K]^T + bias
// One wave computes a 16(M) x 64(N) tile. The k-loop is unrolled by 2 with
// two fragment buffers so the loads for step k+32 are in flight while the
// WMMAs for step k issue (partial s_wait_loadcnt instead of full drains).
// MODE 0: bf16 out as [head][M][64]       (Q / K layouts)
// MODE 1: bf16 out as [head][64][Mtot]    (V transposed)
// MODE 2: f32  out as [M][512]            (final output)
// ---------------------------------------------------------------------------
template <int MODE>
__global__ __launch_bounds__(256) void gemm_bf16_kernel(
    const __bf16* __restrict__ A, const __bf16* __restrict__ Wt,
    const float* __restrict__ bias, int M, int K,
    void* __restrict__ outp, int Mtot) {
  const int lane = threadIdx.x & 31;
  const int ln = lane & 15;
  const int hi = lane >> 4;
  const int wid = blockIdx.x * (blockDim.x >> 5) + (threadIdx.x >> 5);
  const int Mtiles = M >> 4;
  const int mi = wid % Mtiles;
  const int ni = wid / Mtiles;
  if (ni >= 8) return;

  v8f acc[4] = {};
  const __bf16* arow = A + (size_t)(mi * 16 + ln) * K;
  const __bf16* wrow = Wt + (size_t)(ni * 64 + ln) * K;

  BFrag afA, afB, bfA[4], bfB[4];
  load_afrag(afA, arow, hi);
#pragma unroll
  for (int j = 0; j < 4; ++j) load_bfrag(bfA[j], wrow + (size_t)(j * 16) * K, hi);

  // K is a multiple of 64 (K = 512 for every call)
  for (int k0 = 0; k0 < K; k0 += 64) {
    // issue loads for the odd half-step, then compute the even half-step
    load_afrag(afB, arow + k0 + 32, hi);
#pragma unroll
    for (int j = 0; j < 4; ++j)
      load_bfrag(bfB[j], wrow + (size_t)(j * 16) * K + k0 + 32, hi);
    __builtin_prefetch(arow + k0 + 128, 0, 0);   // speculative, 2 steps ahead
#pragma unroll
    for (int j = 0; j < 4; ++j) acc[j] = wmma_bf16(afA, bfA[j], acc[j]);

    if (k0 + 64 < K) {
      load_afrag(afA, arow + k0 + 64, hi);
#pragma unroll
      for (int j = 0; j < 4; ++j)
        load_bfrag(bfA[j], wrow + (size_t)(j * 16) * K + k0 + 64, hi);
    }
#pragma unroll
    for (int j = 0; j < 4; ++j) acc[j] = wmma_bf16(afB, bfB[j], acc[j]);
  }

#pragma unroll
  for (int j = 0; j < 4; ++j) {
    const int n = ni * 64 + j * 16 + ln;
    const float bj = bias[n];
#pragma unroll
    for (int r = 0; r < 8; ++r) {
      const int m = mi * 16 + r + (hi ? 8 : 0);
      const float val = acc[j][r] + bj;
      if (MODE == 0) {
        ((__bf16*)outp)[((size_t)(n >> 6) * M + m) * 64 + (n & 63)] = f2bf(val);
      } else if (MODE == 1) {
        ((__bf16*)outp)[((size_t)(n >> 6) * 64 + (n & 63)) * (size_t)Mtot + m] =
            f2bf(val);
      } else {
        ((float*)outp)[(size_t)m * 512 + n] = val;
      }
    }
  }
}

// ---------------------------------------------------------------------------
// Flash attention, one wave per (head, 16 queries).
// ---------------------------------------------------------------------------
__global__ __launch_bounds__(256) void attn_kernel(
    const __bf16* __restrict__ Qh, const __bf16* __restrict__ Kh,
    const __bf16* __restrict__ Vt, const float2* __restrict__ cmeta,
    const int* __restrict__ batch_x, const int* __restrict__ range,
    __bf16* __restrict__ Ob) {
  const int lane = threadIdx.x & 31;
  const int ln = lane & 15;
  const int hi = lane >> 4;
  const int wid = blockIdx.x * (blockDim.x >> 5) + (threadIdx.x >> 5);
  const int qi = wid & (NX / 16 - 1);
  const int h = wid / (NX / 16);
  const int qbase = qi << 4;

  // Q as B operand of S^T = K @ Q^T : two K-chunks of 32 over DH=64
  const __bf16* qrow = Qh + ((size_t)h * NX + qbase + ln) * DH;
  BFrag qb0, qb1;
  load_bfrag(qb0, qrow, hi);
  load_bfrag(qb1, qrow + 32, hi);

  const int bx = batch_x[qbase + ln];       // this lane's query batch
  const int bmin = batch_x[qbase];          // batch_x sorted -> tile spans
  const int bmax = batch_x[qbase + 15];     // [bmin, bmax]
  // wave-uniform loop bounds -> SGPRs, so loop control is scalar and EXEC
  // stays all-1s around the WMMAs (required by ISA; avoids v_cmpx/exec churn)
  const int clo = __builtin_amdgcn_readfirstlane(range[bmin] & ~31);
  const int chi = __builtin_amdgcn_readfirstlane((range[bmax + 1] + 31) & ~31);

  // strength-reduced strided pointers (advance by one 32-context tile)
  const __bf16*  kptr = Kh + (size_t)h * NC * DH + (size_t)(clo + ln) * DH;
  const __bf16*  vptr = Vt + (size_t)h * DH * NC + (size_t)ln * NC + clo;
  const float2*  cptr = cmeta + clo + (hi ? 8 : 0);

  v8f oacc[4] = {};
  float mrun = -1e30f, lrun = 0.f;

  // prologue: K and V fragments for the first tile
  BFrag k00, k01, k10, k11;
  load_afrag(k00, kptr, hi);
  load_afrag(k01, kptr + 32, hi);
  load_afrag(k10, kptr + 16 * DH, hi);
  load_afrag(k11, kptr + 16 * DH + 32, hi);
  BFrag va0, va1, va2, va3;
  load_afrag(va0, vptr, hi);
  load_afrag(va1, vptr + (size_t)16 * NC, hi);
  load_afrag(va2, vptr + (size_t)32 * NC, hi);
  load_afrag(va3, vptr + (size_t)48 * NC, hi);

  for (int c0 = clo;;) {
    // cmeta for this tile (consumed after the S-WMMAs)
    float2 cm0[8], cm1[8];
#pragma unroll
    for (int r = 0; r < 8; ++r) { cm0[r] = cptr[r]; cm1[r] = cptr[16 + r]; }

    // S^T tiles: ctx rows c0..c0+15 (st0), c0+16..c0+31 (st1); q cols = ln
    v8f st0 = {}, st1 = {};
    st0 = wmma_bf16(k00, qb0, st0);
    st0 = wmma_bf16(k01, qb1, st0);
    st1 = wmma_bf16(k10, qb0, st1);
    st1 = wmma_bf16(k11, qb1, st1);

    // advance + prefetch next tile's K fragments (scalar-uniform branch);
    // softmax below covers their latency
    c0 += 32;
    kptr += (size_t)32 * DH;
    vptr += 32;
    cptr += 32;
    const bool more = (c0 < chi);
    if (more) {
      load_afrag(k00, kptr, hi);
      load_afrag(k01, kptr + 32, hi);
      load_afrag(k10, kptr + 16 * DH, hi);
      load_afrag(k11, kptr + 16 * DH + 32, hi);
      __builtin_prefetch(kptr + (size_t)32 * DH, 0, 0);
      __builtin_prefetch(vptr + 32, 0, 0);
    }

    // frag_mask add + batch/mask gating (C layout: VGPR r -> ctx offset r+8*hi)
    float s0[8], s1[8];
#pragma unroll
    for (int r = 0; r < 8; ++r) {
      s0[r] = (__float_as_int(cm0[r].y) == bx) ? (st0[r] + cm0[r].x) : -1e30f;
      s1[r] = (__float_as_int(cm1[r].y) == bx) ? (st1[r] + cm1[r].x) : -1e30f;
    }

    // online softmax; per-lane stats, halves merged with shfl_xor(16)
    float mloc = -1e30f;
#pragma unroll
    for (int r = 0; r < 8; ++r) mloc = fmaxf(mloc, fmaxf(s0[r], s1[r]));
    mloc = fmaxf(mloc, __shfl_xor(mloc, 16));
    const float mnew = fmaxf(mrun, mloc);
    const float scale = __expf(mrun - mnew);
    float p0[8], p1[8], psum = 0.f;
#pragma unroll
    for (int r = 0; r < 8; ++r) {
      p0[r] = __expf(s0[r] - mnew);
      p1[r] = __expf(s1[r] - mnew);
      psum += p0[r] + p1[r];
    }
    psum += __shfl_xor(psum, 16);
    lrun = lrun * scale + psum;
    mrun = mnew;
#pragma unroll
    for (int d = 0; d < 4; ++d)
#pragma unroll
      for (int r = 0; r < 8; ++r) oacc[d][r] *= scale;

    // Pack P^T into B-operand fragment (lane-pair exchange only):
    //   lo lane needs ctx {0..15} = own st0 ++ partner st0
    //   hi lane needs ctx {16..31} = partner st1 ++ own st1
    BFrag pb;
#pragma unroll
    for (int r = 0; r < 8; ++r) {
      const float t0 = __shfl_xor(p0[r], 16);
      const float t1 = __shfl_xor(p1[r], 16);
      pb.v[r]     = f2bf(hi ? t1 : p0[r]);
      pb.v[8 + r] = f2bf(hi ? p1[r] : t0);
    }

    // O^T += V^T @ P^T  (4 DH-row blocks of 16); V loaded one iteration ago
    oacc[0] = wmma_bf16(va0, pb, oacc[0]);
    oacc[1] = wmma_bf16(va1, pb, oacc[1]);
    oacc[2] = wmma_bf16(va2, pb, oacc[2]);
    oacc[3] = wmma_bf16(va3, pb, oacc[3]);

    if (!more) break;

    // V fragments for the NEXT tile, issued after the O-WMMAs released the
    // registers: a full loop iteration of latency cover before consumption
    load_afrag(va0, vptr, hi);
    load_afrag(va1, vptr + (size_t)16 * NC, hi);
    load_afrag(va2, vptr + (size_t)32 * NC, hi);
    load_afrag(va3, vptr + (size_t)48 * NC, hi);
  }

  const float inv = __builtin_amdgcn_rcpf(lrun);   // v_rcp_f32, ~1 ulp
#pragma unroll
  for (int d = 0; d < 4; ++d)
#pragma unroll
    for (int r = 0; r < 8; ++r) {
      const int dh = 16 * d + r + (hi ? 8 : 0);
      Ob[(size_t)(qbase + ln) * (H * DH) + h * DH + dh] = f2bf(oacc[d][r] * inv);
    }
}

// ---------------------------------------------------------------------------
// Launcher
// ---------------------------------------------------------------------------
extern "C" void kernel_launch(void* const* d_in, const int* in_sizes, int n_in,
                              void* d_out, int out_size, void* d_ws,
                              size_t ws_size, hipStream_t stream) {
  (void)in_sizes; (void)n_in; (void)out_size; (void)ws_size;

  const float* x    = (const float*)d_in[0];
  const float* ctx  = (const float*)d_in[1];
  const int*   bx   = (const int*)d_in[2];
  const int*   bc   = (const int*)d_in[3];
  const unsigned char* mask = (const unsigned char*)d_in[4];  // jax bool = u8
  const float* fm   = (const float*)d_in[5];
  const float* Wqkv = (const float*)d_in[6];
  const float* bqkv = (const float*)d_in[7];
  const float* Wout = (const float*)d_in[8];
  const float* bout = (const float*)d_in[9];

  char* base = (char*)d_ws;
  size_t off = 0;
  auto carve = [&](size_t bytes) -> void* {
    void* r = (void*)(base + off);
    off += (bytes + 255) & ~(size_t)255;
    return r;
  };

  __bf16* xb   = (__bf16*)carve((size_t)NX * DIM * 2);
  __bf16* cb   = (__bf16*)carve((size_t)NC * DIM * 2);
  __bf16* Wqt  = (__bf16*)carve((size_t)512 * 512 * 2);
  __bf16* Wkt  = (__bf16*)carve((size_t)512 * 512 * 2);
  __bf16* Wvt  = (__bf16*)carve((size_t)512 * 512 * 2);
  __bf16* Wot  = (__bf16*)carve((size_t)512 * 512 * 2);
  __bf16* Qh   = (__bf16*)carve((size_t)H * NX * DH * 2);
  __bf16* Kh   = (__bf16*)carve((size_t)H * NC * DH * 2);
  __bf16* Vt   = (__bf16*)carve((size_t)H * DH * NC * 2);
  __bf16* Ob   = (__bf16*)carve((size_t)NX * H * DH * 2);
  float2* cmeta = (float2*)carve((size_t)NC * sizeof(float2));
  int*    range = (int*)carve((size_t)(BATCHES + 1) * sizeof(int));

  // 1. bf16 operand copies (vectorized x4)
  cvt_bf16_kernel<<<(NX * DIM / 4 + 255) / 256, 256, 0, stream>>>(x, xb, NX * DIM / 4);
  cvt_bf16_kernel<<<(NC * DIM / 4 + 255) / 256, 256, 0, stream>>>(ctx, cb, NC * DIM / 4);

  // 2. weight transposes (K-major bf16)
  transpose_w_kernel<<<(512 * 512 + 255) / 256, 256, 0, stream>>>(Wqkv, 1536, 0,    Wqt);
  transpose_w_kernel<<<(512 * 512 + 255) / 256, 256, 0, stream>>>(Wqkv, 1536, 512,  Wkt);
  transpose_w_kernel<<<(512 * 512 + 255) / 256, 256, 0, stream>>>(Wqkv, 1536, 1024, Wvt);
  transpose_w_kernel<<<(512 * 512 + 255) / 256, 256, 0, stream>>>(Wout, 512,  0,    Wot);

  // 3. context metadata + batch ranges
  prep_cmeta_kernel<<<(NC + 255) / 256, 256, 0, stream>>>(bc, mask, fm, cmeta, NC);
  batch_ranges_kernel<<<1, 32, 0, stream>>>(bc, range);

  // 4. projections (16x64 tile per wave; 8 N-strips)
  const int gq = (NX / 16) * 8 * 32 / 256;   // 128 blocks
  const int gc = (NC / 16) * 8 * 32 / 256;   // 512 blocks
  gemm_bf16_kernel<0><<<gq, 256, 0, stream>>>(xb, Wqt, bqkv,        NX, DIM, Qh, 0);
  gemm_bf16_kernel<0><<<gc, 256, 0, stream>>>(cb, Wkt, bqkv + 512,  NC, DIM, Kh, 0);
  gemm_bf16_kernel<1><<<gc, 256, 0, stream>>>(cb, Wvt, bqkv + 1024, NC, DIM, Vt, NC);

  // 5. flash attention: H * NX/16 = 1024 waves -> 128 blocks
  attn_kernel<<<H * (NX / 16) * 32 / 256, 256, 0, stream>>>(Qh, Kh, Vt, cmeta,
                                                            bx, range, Ob);

  // 6. output projection -> f32 d_out
  gemm_bf16_kernel<2><<<gq, 256, 0, stream>>>(Ob, Wot, bout, NX, DIM, d_out, 0);
}
